// Network_57861799412252
// MI455X (gfx1250) — matrix operand order repaired and checked by reading, not compiled, not run on hardware
//
#include <hip/hip_runtime.h>
#include <hip/hip_bf16.h>

typedef _Float16 h4  __attribute__((ext_vector_type(4)));
typedef _Float16 h8  __attribute__((ext_vector_type(8)));
typedef _Float16 v16h __attribute__((ext_vector_type(16)));
typedef float    f4  __attribute__((ext_vector_type(4)));
typedef float    v8f __attribute__((ext_vector_type(8)));

// ---------------------------------------------------------------------------
// WMMA 1x1-conv GEMM. Per sample b: OUT[co, n] = sum_ci W[co,ci] * IN[ci, n].
// IN/OUT f16 [B, C, HWs], HWs % 16 == 0 (padded stride -> no N guards).
// W f32 [Cout, Cin], Cin <= 120. 128 threads = 4 wave32, one 16x16 C tile/wave.
//  - A panel (16 x Kpad) + pre-combined norm scale/bias preloaded to LDS once;
//    K loop has NO barriers (sA/sS/sT read-only, sB wave-private).
//  - Full K steps are completely guard-free; single guarded tail step.
//  - Optional fused input instance-norm+ReLU folded to one fma+max per element
//    against LDS-resident scale/bias.
// ---------------------------------------------------------------------------
__global__ __launch_bounds__(128)
void gemm1x1_wmma(const _Float16* __restrict__ in, const float* __restrict__ w,
                  _Float16* __restrict__ out,
                  const float* __restrict__ gN, const float* __restrict__ bN,
                  const float* __restrict__ mN, const float* __restrict__ ivN,
                  const int* __restrict__ dev_ids,
                  int Cin, int Cout, int HWs)
{
  __shared__ __align__(16) _Float16 sA[16 * 128];        // [m][Kpad]
  __shared__ __align__(16) _Float16 sB[2][4][16][40];    // ping-pong, per-wave
  __shared__ float sS[128], sT[128];                     // norm scale/bias

  const int b   = blockIdx.z;
  const int m0  = blockIdx.y * 16;
  const int tid = threadIdx.x;
  const int wv  = tid >> 5;
  const int ln  = tid & 31;
  int n0 = blockIdx.x * 64 + wv * 16;
  if (n0 > HWs - 16) n0 = HWs - 16;     // clamp: duplicate tile, benign

  const int  Kpad    = (Cin + 31) & ~31;   // <= 128
  const bool useNorm = (gN != nullptr);
  const int  d       = dev_ids ? dev_ids[b] : 0;

  // ---- preload A panel once (guards hoisted out of the hot loop) ----
  for (int q = tid; q < 16 * (Kpad >> 2); q += 128) {
    int r  = q / (Kpad >> 2);
    int kq = (q % (Kpad >> 2)) * 4;
    h4 hv = {};
    if (m0 + r < Cout) {
#pragma unroll
      for (int j = 0; j < 4; ++j)
        if (kq + j < Cin) hv[j] = (_Float16)w[(size_t)(m0 + r) * Cin + kq + j];
    }
    *(h4*)&sA[r * Kpad + kq] = hv;
  }
  // ---- pre-combine norm params: y = scale*x + bias (zero-padded to 128) ----
  if (useNorm) {
    int ck = tid;
    if (ck < 128) {
      float sc = 0.f, bi = 0.f;
      if (ck < Cin) {
        sc = gN[d * Cin + ck] * ivN[b * Cin + ck];
        bi = bN[d * Cin + ck] - mN[b * Cin + ck] * sc;
      }
      sS[ck] = sc; sT[ck] = bi;
    }
  }
  __syncthreads();                       // the only block-wide barrier

  const _Float16* inb  = in  + (size_t)b * Cin  * HWs;
  _Float16*       outb = out + (size_t)b * Cout * HWs;

  v8f acc = {};

  // fragment geometry (CDNA5 16x16x32 f16 layouts)
  const int mA = ln & 15;
  const int kA = (ln < 16) ? 0 : 8;
  const int nB = ln & 15;
  const int kB = (ln < 16) ? 0 : 8;
  // staging geometry: 64 chunks of 8 halves cover 32k x 16n
  const int c0k = ln >> 1,        c0o = (ln & 1) * 8;        // chunk ln
  const int c1k = (32 + ln) >> 1, c1o = ((32 + ln) & 1) * 8; // chunk 32+ln

  const int fullSteps = Cin >> 5;
  const int kRem      = Cin & 31;

  int p = 0, k0 = 0;
  for (int s = 0; s < fullSteps; ++s, k0 += 32, p ^= 1) {
    // ---- stage B: guard-free coalesced b128 loads, wave-private scatter ----
    h8 v0 = *(const h8*)&inb[(size_t)(k0 + c0k) * HWs + n0 + c0o];
    h8 v1 = *(const h8*)&inb[(size_t)(k0 + c1k) * HWs + n0 + c1o];
    if (useNorm) {
      float sc0 = sS[k0 + c0k], bi0 = sT[k0 + c0k];
      float sc1 = sS[k0 + c1k], bi1 = sT[k0 + c1k];
      h8 o0, o1;
#pragma unroll
      for (int j = 0; j < 8; ++j) {
        o0[j] = (_Float16)fmaxf(fmaf((float)v0[j], sc0, bi0), 0.f);
        o1[j] = (_Float16)fmaxf(fmaf((float)v1[j], sc1, bi1), 0.f);
      }
      v0 = o0; v1 = o1;
    }
#pragma unroll
    for (int j = 0; j < 8; ++j) {
      sB[p][wv][c0o + j][c0k] = v0[j];
      sB[p][wv][c1o + j][c1k] = v1[j];
    }
    if (k0 + 32 < Cin)
      __builtin_prefetch(&inb[(size_t)(k0 + 32) * HWs + n0], 0, 1);

    v16h a, bb;
    {
      h8 lo = *(const h8*)&sA[mA * Kpad + k0 + kA];
      h8 hi = *(const h8*)&sA[mA * Kpad + k0 + kA + 16];
#pragma unroll
      for (int i = 0; i < 8; ++i) { a[i] = lo[i]; a[i + 8] = hi[i]; }
    }
    {
      h8 lo = *(const h8*)&sB[p][wv][nB][kB];
      h8 hi = *(const h8*)&sB[p][wv][nB][kB + 16];
#pragma unroll
      for (int i = 0; i < 8; ++i) { bb[i] = lo[i]; bb[i + 8] = hi[i]; }
    }
    acc = __builtin_amdgcn_wmma_f32_16x16x32_f16(false, a, false, bb,
                                                 (short)0, acc, false, false);
  }

  if (kRem) {                            // single guarded tail step
    h8 v0 = {}, v1 = {};
    if (c0k < kRem) v0 = *(const h8*)&inb[(size_t)(k0 + c0k) * HWs + n0 + c0o];
    if (c1k < kRem) v1 = *(const h8*)&inb[(size_t)(k0 + c1k) * HWs + n0 + c1o];
    if (useNorm) {                       // sS/sT zero-padded: no guard needed
      float sc0 = sS[k0 + c0k], bi0 = sT[k0 + c0k];
      float sc1 = sS[k0 + c1k], bi1 = sT[k0 + c1k];
      h8 o0, o1;
#pragma unroll
      for (int j = 0; j < 8; ++j) {
        o0[j] = (_Float16)fmaxf(fmaf((float)v0[j], sc0, bi0), 0.f);
        o1[j] = (_Float16)fmaxf(fmaf((float)v1[j], sc1, bi1), 0.f);
      }
      v0 = o0; v1 = o1;
    }
#pragma unroll
    for (int j = 0; j < 8; ++j) {
      sB[p][wv][c0o + j][c0k] = v0[j];
      sB[p][wv][c1o + j][c1k] = v1[j];
    }
    v16h a, bb;
    {
      h8 lo = *(const h8*)&sA[mA * Kpad + k0 + kA];
      h8 hi = *(const h8*)&sA[mA * Kpad + k0 + kA + 16];
#pragma unroll
      for (int i = 0; i < 8; ++i) { a[i] = lo[i]; a[i + 8] = hi[i]; }
    }
    {
      h8 lo = *(const h8*)&sB[p][wv][nB][kB];
      h8 hi = *(const h8*)&sB[p][wv][nB][kB + 16];
#pragma unroll
      for (int i = 0; i < 8; ++i) { bb[i] = lo[i]; bb[i + 8] = hi[i]; }
    }
    acc = __builtin_amdgcn_wmma_f32_16x16x32_f16(false, a, false, bb,
                                                 (short)0, acc, false, false);
  }

  // C/D layout: VGPR r -> M = r (lanes 0-15) / 8+r (lanes 16-31), N = lane&15
  const int rbase = (ln < 16) ? 0 : 8;
  const int n = n0 + nB;
  if (m0 + 16 <= Cout) {
#pragma unroll
    for (int r = 0; r < 8; ++r)
      outb[(size_t)(m0 + rbase + r) * HWs + n] = (_Float16)acc[r];
  } else {
#pragma unroll
    for (int r = 0; r < 8; ++r) {
      int m = m0 + rbase + r;
      if (m < Cout) outb[(size_t)m * HWs + n] = (_Float16)acc[r];
    }
  }
}

// ---------------------------------------------------------------------------
// Stem convs (tiny channel counts -> direct VALU)
// ---------------------------------------------------------------------------
__global__ void stem0_conv(const float* __restrict__ x, const float* __restrict__ w,
                           _Float16* __restrict__ out, int total)
{ // [B,1,256,128] -> [B,8,127,63] (stride 8001, unpadded), 3x3 s2 p0, ReLU
  int i = blockIdx.x * blockDim.x + threadIdx.x;
  if (i >= total) return;
  int wo = i % 63; int t = i / 63;
  int ho = t % 127; t /= 127;
  int c = t % 8; int b = t / 8;
  const float* xb = x + (size_t)b * 256 * 128;
  float acc = 0.f;
#pragma unroll
  for (int kh = 0; kh < 3; ++kh)
#pragma unroll
    for (int kw = 0; kw < 3; ++kw)
      acc += w[c * 9 + kh * 3 + kw] * xb[(2 * ho + kh) * 128 + (2 * wo + kw)];
  out[i] = (_Float16)fmaxf(acc, 0.f);
}

__global__ void stem1_conv(const _Float16* __restrict__ a0, const float* __restrict__ w,
                           _Float16* __restrict__ out, int total)
{ // [B,8,127,63] -> [B,32,63x31] stored with stride 1968, 3x3 s2 p0, ReLU
  int i = blockIdx.x * blockDim.x + threadIdx.x;
  if (i >= total) return;
  int wo = i % 31; int t = i / 31;
  int ho = t % 63; t /= 63;
  int c = t % 32; int b = t / 32;
  const _Float16* ab = a0 + (size_t)b * 8 * 127 * 63;
  float acc = 0.f;
  for (int ci = 0; ci < 8; ++ci)
#pragma unroll
    for (int kh = 0; kh < 3; ++kh)
#pragma unroll
      for (int kw = 0; kw < 3; ++kw)
        acc += w[((c * 8 + ci) * 3 + kh) * 3 + kw] *
               (float)ab[((size_t)ci * 127 + (2 * ho + kh)) * 63 + (2 * wo + kw)];
  out[((size_t)b * 32 + c) * 1968 + ho * 31 + wo] = (_Float16)fmaxf(acc, 0.f);
}

// ---------------------------------------------------------------------------
// Depthwise 3x3 (pad 1) with FUSED input instance-norm + ReLU (expand BN).
// ---------------------------------------------------------------------------
__global__ void dwconv3x3_bn(const _Float16* __restrict__ in, const float* __restrict__ w,
                             const float* __restrict__ gN, const float* __restrict__ bN,
                             const float* __restrict__ mN, const float* __restrict__ ivN,
                             const int* __restrict__ dev_ids,
                             _Float16* __restrict__ out, int C, int H, int W,
                             int sh, int sw, int Ho, int Wo, int sIn, int sOut, int total)
{
  int i = blockIdx.x * blockDim.x + threadIdx.x;
  if (i >= total) return;
  int wo = i % Wo; int t = i / Wo;
  int ho = t % Ho; t /= Ho;
  int c = t % C; int b = t / C;
  int d = dev_ids[b];
  float sc = gN[d * C + c] * ivN[b * C + c];
  float bi = bN[d * C + c] - mN[b * C + c] * sc;
  const _Float16* ib = in + ((size_t)b * C + c) * sIn;
  float acc = 0.f;
#pragma unroll
  for (int kh = 0; kh < 3; ++kh) {
    int hi = ho * sh - 1 + kh;
    if (hi < 0 || hi >= H) continue;
#pragma unroll
    for (int kw = 0; kw < 3; ++kw) {
      int wi = wo * sw - 1 + kw;
      if (wi < 0 || wi >= W) continue;
      float xv = fmaxf(fmaf((float)ib[hi * W + wi], sc, bi), 0.f);
      acc += w[c * 9 + kh * 3 + kw] * xv;
    }
  }
  out[((size_t)b * C + c) * sOut + ho * Wo + wo] = (_Float16)acc;
}

__global__ void avgpool3(const _Float16* __restrict__ in, _Float16* __restrict__ out,
                         int C, int H, int W, int sh, int sw, int Ho, int Wo,
                         int sIn, int sOut, int total)
{
  int i = blockIdx.x * blockDim.x + threadIdx.x;
  if (i >= total) return;
  int wo = i % Wo; int t = i / Wo;
  int ho = t % Ho; t /= Ho;
  int c = t % C; int b = t / C;
  const _Float16* ib = in + ((size_t)b * C + c) * sIn;
  float acc = 0.f;
#pragma unroll
  for (int kh = 0; kh < 3; ++kh) {
    int hi = ho * sh - 1 + kh;
    if (hi < 0 || hi >= H) continue;
#pragma unroll
    for (int kw = 0; kw < 3; ++kw) {
      int wi = wo * sw - 1 + kw;
      if (wi < 0 || wi >= W) continue;
      acc += (float)ib[hi * W + wi];
    }
  }
  out[((size_t)b * C + c) * sOut + ho * Wo + wo] = (_Float16)(acc * (1.f / 9.f));
}

// ---------------------------------------------------------------------------
// Instance-norm stats per (b,c): mean + rsqrt(var+eps), unbiased (ddof=1).
// ---------------------------------------------------------------------------
__global__ __launch_bounds__(256)
void bn_stats(const _Float16* __restrict__ x, float* __restrict__ mean,
              float* __restrict__ inv, int HW, int HWs)
{
  int bc = blockIdx.x;
  const _Float16* p = x + (size_t)bc * HWs;
  float s1 = 0.f, s2 = 0.f;
  int nv = HW >> 3;
  for (int t = threadIdx.x; t < nv; t += 256) {
    h8 v = *(const h8*)&p[t * 8];
#pragma unroll
    for (int j = 0; j < 8; ++j) { float f = (float)v[j]; s1 += f; s2 += f * f; }
  }
  for (int t = (nv << 3) + threadIdx.x; t < HW; t += 256) {
    float f = (float)p[t]; s1 += f; s2 += f * f;
  }
  __shared__ float r1[256], r2[256];
  r1[threadIdx.x] = s1; r2[threadIdx.x] = s2;
  __syncthreads();
  for (int s = 128; s > 0; s >>= 1) {
    if (threadIdx.x < s) { r1[threadIdx.x] += r1[threadIdx.x + s];
                           r2[threadIdx.x] += r2[threadIdx.x + s]; }
    __syncthreads();
  }
  if (threadIdx.x == 0) {
    float m = r1[0] / (float)HW;
    float var = fmaxf((r2[0] - r1[0] * m) / (float)(HW - 1), 0.f);
    mean[bc] = m;
    inv[bc]  = rsqrtf(var + 1e-5f);
  }
}

// y = [relu]( scale*x + bias [+ add] ), in place, h8-vectorized.
__global__ void bn_apply_v8(_Float16* __restrict__ x, const float* __restrict__ gamma,
                            const float* __restrict__ beta, const float* __restrict__ mean,
                            const float* __restrict__ inv, const int* __restrict__ dev_ids,
                            const _Float16* __restrict__ add, int C, int HWs,
                            int totalv, int relu)
{
  int i = blockIdx.x * blockDim.x + threadIdx.x;
  if (i >= totalv) return;
  int vc = HWs >> 3;
  int c = (i / vc) % C;
  int b = i / (vc * C);
  int d = dev_ids[b];
  size_t base = ((size_t)b * C + c) * HWs + (size_t)(i % vc) * 8;
  float sc = gamma[d * C + c] * inv[b * C + c];
  float bi = beta[d * C + c] - mean[b * C + c] * sc;
  h8 v = *(const h8*)&x[base];
  h8 o;
  if (add) {
    h8 av = *(const h8*)&add[base];
#pragma unroll
    for (int j = 0; j < 8; ++j) {
      float y = fmaf((float)v[j], sc, bi) + (float)av[j];
      if (relu) y = fmaxf(y, 0.f);
      o[j] = (_Float16)y;
    }
  } else {
#pragma unroll
    for (int j = 0; j < 8; ++j) {
      float y = fmaf((float)v[j], sc, bi);
      if (relu) y = fmaxf(y, 0.f);
      o[j] = (_Float16)y;
    }
  }
  *(h8*)&x[base] = o;
}

// out[b,k] = mean_HW( gamma*(x - mean)*inv + beta ), HW = 256, K = 10
__global__ __launch_bounds__(256)
void final_mean(const _Float16* __restrict__ x, const float* __restrict__ gamma,
                const float* __restrict__ beta, const float* __restrict__ mean,
                const float* __restrict__ inv, const int* __restrict__ dev_ids,
                float* __restrict__ out)
{
  int bk = blockIdx.x; int k = bk % 10; int b = bk / 10;
  const _Float16* p = x + ((size_t)b * 10 + k) * 256;
  __shared__ float r[256];
  r[threadIdx.x] = (float)p[threadIdx.x];
  __syncthreads();
  for (int s = 128; s > 0; s >>= 1) {
    if (threadIdx.x < s) r[threadIdx.x] += r[threadIdx.x + s];
    __syncthreads();
  }
  if (threadIdx.x == 0) {
    int d = dev_ids[b];
    out[bk] = gamma[d * 10 + k] * ((r[0] / 256.f) - mean[b * 10 + k]) * inv[b * 10 + k]
            + beta[d * 10 + k];
  }
}

// ---------------------------------------------------------------------------
// Host orchestration
// ---------------------------------------------------------------------------
extern "C" void kernel_launch(void* const* d_in, const int* in_sizes, int n_in,
                              void* d_out, int out_size, void* d_ws, size_t ws_size,
                              hipStream_t stream)
{
  (void)in_sizes; (void)n_in; (void)out_size; (void)ws_size;
  const float* x     = (const float*)d_in[0];
  const int*   devid = (const int*)d_in[1];
  auto P = [&](int i) { return (const float*)d_in[i]; };

  char* ws = (char*)d_ws;
  const size_t MB = 1024 * 1024;
  _Float16* H0  = (_Float16*)(ws);             // 32 MB  ping
  _Float16* H1  = (_Float16*)(ws + 32 * MB);   // 32 MB  pong
  _Float16* E   = (_Float16*)(ws + 64 * MB);   // 64 MB  expand (also stem A0)
  _Float16* D   = (_Float16*)(ws + 128 * MB);  // 64 MB  depthwise
  _Float16* SC  = (_Float16*)(ws + 192 * MB);  // 16 MB  shortcut (also ff out)
  _Float16* SC2 = (_Float16*)(ws + 208 * MB);  // 8 MB   shortcut pool temp
  float* MEAN0  = (float*)(ws + 216 * MB);     // stats ping
  float* INV0   = (float*)(ws + 217 * MB);
  float* MEAN1  = (float*)(ws + 218 * MB);     // stats pong (proj output)
  float* INV1   = (float*)(ws + 219 * MB);

  const int B = 256;
  auto cdiv = [](int a, int b) { return (a + b - 1) / b; };
  const float* NUL = nullptr;

  // ---- stem ----
  {
    int t0 = B * 8 * 127 * 63;
    stem0_conv<<<cdiv(t0, 256), 256, 0, stream>>>(x, P(2), E, t0);
    int t1 = B * 32 * 63 * 31;
    stem1_conv<<<cdiv(t1, 256), 256, 0, stream>>>(E, P(3), H0, t1);
  }

  struct Blk { int pb, cin, cout, expc, H, W, sh, sw, Ho, Wo, has_sc, sIn, sOut; };
  const Blk blks[6] = {
    { 4, 32,  32,  64, 63, 31, 1, 1, 63, 31, 0, 1968, 1968},
    {13, 32,  32,  64, 63, 31, 2, 2, 32, 16, 0, 1968,  512},
    {22, 32,  32,  64, 32, 16, 1, 1, 32, 16, 0,  512,  512},
    {31, 32,  56,  64, 32, 16, 2, 1, 16, 16, 1,  512,  256},
    {43, 56,  56, 120, 16, 16, 1, 1, 16, 16, 0,  256,  256},
    {52, 56, 104, 120, 16, 16, 1, 1, 16, 16, 1,  256,  256},
  };

  _Float16* hb = H0;
  _Float16* nb = H1;
  for (int i = 0; i < 6; ++i) {
    const Blk& k = blks[i];
    const int HWin = k.H * k.W, HWout = k.Ho * k.Wo;
    const int sIn = k.sIn, sOut = k.sOut;

    // 1) expand 1x1 (WMMA, raw out) + stats
    gemm1x1_wmma<<<dim3(cdiv(sIn, 64), cdiv(k.expc, 16), B), 128, 0, stream>>>(
        hb, P(k.pb + 0), E, NUL, NUL, NUL, NUL, nullptr, k.cin, k.expc, sIn);
    bn_stats<<<B * k.expc, 256, 0, stream>>>(E, MEAN0, INV0, HWin, sIn);

    // 2) depthwise 3x3 with fused expand-norm+ReLU on input (raw out) + stats
    int tD = B * k.expc * HWout;
    dwconv3x3_bn<<<cdiv(tD, 256), 256, 0, stream>>>(E, P(k.pb + 3),
        P(k.pb + 1), P(k.pb + 2), MEAN0, INV0, devid, D, k.expc,
        k.H, k.W, k.sh, k.sw, k.Ho, k.Wo, sIn, sOut, tD);
    bn_stats<<<B * k.expc, 256, 0, stream>>>(D, MEAN0, INV0, HWout, sOut);

    // 3) project 1x1 (WMMA) with fused dw-norm+ReLU on input (raw out) + stats
    gemm1x1_wmma<<<dim3(cdiv(sOut, 64), cdiv(k.cout, 16), B), 128, 0, stream>>>(
        D, P(k.pb + 6), nb, P(k.pb + 4), P(k.pb + 5), MEAN0, INV0, devid,
        k.expc, k.cout, sOut);
    bn_stats<<<B * k.cout, 256, 0, stream>>>(nb, MEAN1, INV1, HWout, sOut);

    // 4) shortcut
    const _Float16* scp = nullptr;
    if (!k.has_sc) {
      if (k.sh == 1 && k.sw == 1) {
        scp = hb;
      } else {
        int tP = B * k.cin * HWout;
        avgpool3<<<cdiv(tP, 256), 256, 0, stream>>>(hb, SC, k.cin,
            k.H, k.W, k.sh, k.sw, k.Ho, k.Wo, sIn, sOut, tP);
        scp = SC;
      }
    } else {
      const _Float16* src = hb;
      int srcStride = sIn;
      if (k.sh != 1 || k.sw != 1) {
        int tP = B * k.cin * HWout;
        avgpool3<<<cdiv(tP, 256), 256, 0, stream>>>(hb, SC2, k.cin,
            k.H, k.W, k.sh, k.sw, k.Ho, k.Wo, sIn, sOut, tP);
        src = SC2; srcStride = sOut;
      }
      gemm1x1_wmma<<<dim3(cdiv(srcStride, 64), cdiv(k.cout, 16), B), 128, 0, stream>>>(
          src, P(k.pb + 9), SC, NUL, NUL, NUL, NUL, nullptr, k.cin, k.cout, srcStride);
      bn_stats<<<B * k.cout, 256, 0, stream>>>(SC, MEAN0, INV0, HWout, sOut);
      int tSv = B * k.cout * (sOut >> 3);
      bn_apply_v8<<<cdiv(tSv, 256), 256, 0, stream>>>(SC, P(k.pb + 10), P(k.pb + 11),
          MEAN0, INV0, devid, nullptr, k.cout, sOut, tSv, 0);
      scp = SC;
    }

    // 5) project norm + residual add + ReLU -> next h
    int tOv = B * k.cout * (sOut >> 3);
    bn_apply_v8<<<cdiv(tOv, 256), 256, 0, stream>>>(nb, P(k.pb + 7), P(k.pb + 8),
        MEAN1, INV1, devid, scp, k.cout, sOut, tOv, 1);

    _Float16* tmp = hb; hb = nb; nb = tmp;
  }

  // ---- final 1x1 (104->10, WMMA) + IN-norm + global mean ----
  gemm1x1_wmma<<<dim3(cdiv(256, 64), 1, B), 128, 0, stream>>>(
      hb, P(64), SC, NUL, NUL, NUL, NUL, nullptr, 104, 10, 256);
  bn_stats<<<B * 10, 256, 0, stream>>>(SC, MEAN0, INV0, 256, 256);
  final_mean<<<B * 10, 256, 0, stream>>>(SC, P(65), P(66), MEAN0, INV0, devid, (float*)d_out);
}
